// Model_22265110462499
// MI455X (gfx1250) — matrix operand order repaired and checked by reading, not compiled, not run on hardware
//
#include <hip/hip_runtime.h>

typedef __attribute__((ext_vector_type(16))) _Float16 v16h;
typedef __attribute__((ext_vector_type(8)))  _Float16 v8h;
typedef __attribute__((ext_vector_type(4)))  _Float16 v4h;
typedef __attribute__((ext_vector_type(8)))  float    v8f;

#define M_DIM 4096
#define K_DIM 4096
#define N_DIM 11008

#define BM 128
#define BN 128
#define BK 32
#define LDSH 40   // LDS row pitch in halves (32 data + 8 pad -> 80B, 16B aligned)

// ---------------------------------------------------------------------------
// rowsum[m] = sum_k x[m,k]   (folds the dequant offset into the epilogue)
// ---------------------------------------------------------------------------
__global__ __launch_bounds__(256) void rowsum_kernel(const float* __restrict__ x,
                                                     float* __restrict__ rsum) {
    __shared__ float red[256];
    const int row = blockIdx.x;
    const float* xr = x + (size_t)row * K_DIM;
    float s = 0.f;
    for (int k = threadIdx.x; k < K_DIM; k += 256) s += xr[k];
    red[threadIdx.x] = s;
    __syncthreads();
    for (int off = 128; off > 0; off >>= 1) {
        if (threadIdx.x < off) red[threadIdx.x] += red[threadIdx.x + off];
        __syncthreads();
    }
    if (threadIdx.x == 0) rsum[row] = red[0];
}

// ---------------------------------------------------------------------------
// One-time pre-pass: x f32 -> f16 (row-major, unchanged layout)
// ---------------------------------------------------------------------------
__global__ __launch_bounds__(256) void convert_x_kernel(const float* __restrict__ x,
                                                        _Float16* __restrict__ xh) {
    const size_t base = ((size_t)blockIdx.x * 256 + threadIdx.x) * 8;
    const float4 a = *(const float4*)(x + base);
    const float4 b = *(const float4*)(x + base + 4);
    v8h h = { (_Float16)a.x, (_Float16)a.y, (_Float16)a.z, (_Float16)a.w,
              (_Float16)b.x, (_Float16)b.y, (_Float16)b.z, (_Float16)b.w };
    *(v8h*)(xh + base) = h;
}

// ---------------------------------------------------------------------------
// One-time pre-pass: w int32 [K,N] -> f16 transposed [N,K] (K contiguous).
// 64x64 tile through LDS; read coalesced over N, write coalesced over K.
// Converts once instead of once per row-block (32x reuse downstream).
// ---------------------------------------------------------------------------
__global__ __launch_bounds__(256) void transpose_w_kernel(const int* __restrict__ w,
                                                          _Float16* __restrict__ whT) {
    __shared__ __align__(16) _Float16 tile[64 * 80];   // pitch 80 halves = 160B
    const int tid = threadIdx.x;
    const int kt = blockIdx.x * 64;
    const int nt = blockIdx.y * 64;
    #pragma unroll
    for (int i = 0; i < 4; ++i) {
        const int idx = tid + i * 256;       // 0..1023 int4 slots (64 k-rows x 16)
        const int k  = idx >> 4;
        const int c4 = idx & 15;
        const int4 v = *(const int4*)(w + (size_t)(kt + k) * N_DIM + nt + c4 * 4);
        tile[(c4 * 4 + 0) * 80 + k] = (_Float16)(float)v.x;
        tile[(c4 * 4 + 1) * 80 + k] = (_Float16)(float)v.y;
        tile[(c4 * 4 + 2) * 80 + k] = (_Float16)(float)v.z;
        tile[(c4 * 4 + 3) * 80 + k] = (_Float16)(float)v.w;
    }
    __syncthreads();
    #pragma unroll
    for (int i = 0; i < 2; ++i) {
        const int idx = tid + i * 256;       // 0..511 v8h chunks (64 n-rows x 8)
        const int n  = idx >> 3;
        const int c8 = idx & 7;
        v8h h = *(const v8h*)&tile[n * 80 + c8 * 8];
        *(v8h*)(whT + (size_t)(nt + n) * K_DIM + kt + c8 * 8) = h;
    }
}

// ---------------------------------------------------------------------------
// CDNA5 async copy: global -> LDS, 16B per lane, tracked by ASYNCcnt
// ---------------------------------------------------------------------------
__device__ __forceinline__ void async_copy_b128(const _Float16* gsrc, _Float16* ldst) {
    unsigned lds_off = (unsigned)(uintptr_t)ldst;   // low 32 bits = LDS address
    asm volatile("global_load_async_to_lds_b128 %0, %1, off"
                 :: "v"(lds_off), "v"(gsrc) : "memory");
}
__device__ __forceinline__ void async_wait_and_barrier() {
    asm volatile("s_wait_asynccnt 0x0" ::: "memory");
    __syncthreads();
}

// stage a 128-row x 32-half tile (K contiguous in global) into LDS, async
__device__ __forceinline__ void stage_tile_async(const _Float16* __restrict__ gbase,
                                                 int rowBase, int kk,
                                                 _Float16* __restrict__ lds, int tid) {
    #pragma unroll
    for (int i = 0; i < 2; ++i) {
        const int idx = tid + i * 256;   // 0..511 b128 chunks (128 rows x 4)
        const int row = idx >> 2;
        const int c   = idx & 3;
        async_copy_b128(gbase + (size_t)(rowBase + row) * K_DIM + kk + c * 8,
                        lds + row * LDSH + c * 8);
    }
}

// fragment loads (per ISA 16-bit A/B layouts) + 8 WMMAs for one 32-K step
__device__ __forceinline__ void wmma_step(const _Float16* __restrict__ Asb,
                                          const _Float16* __restrict__ Bsb,
                                          int waveM, int waveN, int hi, int l15,
                                          v8f acc[4][2]) {
    v16h afrag[4];
    #pragma unroll
    for (int mt = 0; mt < 4; ++mt) {
        const _Float16* p = &Asb[(waveM * 64 + mt * 16 + l15) * LDSH];
        v8h a0 = *(const v8h*)(p + hi * 8);
        v8h a1 = *(const v8h*)(p + 16 + hi * 8);
        afrag[mt] = __builtin_shufflevector(a0, a1, 0,1,2,3,4,5,6,7,8,9,10,11,12,13,14,15);
    }
    v16h bfrag[2];
    #pragma unroll
    for (int nt = 0; nt < 2; ++nt) {
        const _Float16* p = &Bsb[(waveN * 32 + nt * 16 + l15) * LDSH];
        v8h b0 = *(const v8h*)(p + hi * 8);
        v8h b1 = *(const v8h*)(p + 16 + hi * 8);
        bfrag[nt] = __builtin_shufflevector(b0, b1, 0,1,2,3,4,5,6,7,8,9,10,11,12,13,14,15);
    }
    #pragma unroll
    for (int mt = 0; mt < 4; ++mt)
        #pragma unroll
        for (int nt = 0; nt < 2; ++nt)
            acc[mt][nt] = __builtin_amdgcn_wmma_f32_16x16x32_f16(
                false, afrag[mt], false, bfrag[nt], (short)0, acc[mt][nt], false, false);
}

// ---------------------------------------------------------------------------
// Fast GEMM: pre-converted fp16 operands, double-buffered LDS, async staging.
// 256 threads = 8 waves; 128x128 tile; each wave 64x32 (4x2 WMMA tiles).
// ---------------------------------------------------------------------------
__global__ __launch_bounds__(256) void dequant_gemm_f16_kernel(
    const _Float16* __restrict__ xh, const _Float16* __restrict__ whT,
    const float* __restrict__ scale, const float* __restrict__ offs,
    const float* __restrict__ rsum, float* __restrict__ out) {

    __shared__ __align__(16) _Float16 As[2][BM * LDSH];
    __shared__ __align__(16) _Float16 Bs[2][BN * LDSH];

    const int tid  = threadIdx.x;
    const int lane = tid & 31;
    const int wave = tid >> 5;
    const int hi   = lane >> 4;
    const int l15  = lane & 15;
    const int waveM = wave >> 2;   // 0..1
    const int waveN = wave & 3;    // 0..3

    const int rowBase0 = blockIdx.x * BM;   // row-blocks fastest -> weight hot in L2
    const int colBase0 = blockIdx.y * BN;

    v8f acc[4][2];
    #pragma unroll
    for (int i = 0; i < 4; ++i)
        #pragma unroll
        for (int j = 0; j < 2; ++j)
            acc[i][j] = v8f{};

    // prologue: stage first tile into buffer 0
    stage_tile_async(xh,  rowBase0, 0, As[0], tid);
    stage_tile_async(whT, colBase0, 0, Bs[0], tid);
    async_wait_and_barrier();

    for (int kk = 0; kk < K_DIM; kk += 2 * BK) {
        // stage kk+32 into buf1 while computing buf0
        if (kk + BK < K_DIM) {
            stage_tile_async(xh,  rowBase0, kk + BK, As[1], tid);
            stage_tile_async(whT, colBase0, kk + BK, Bs[1], tid);
        }
        wmma_step(As[0], Bs[0], waveM, waveN, hi, l15, acc);
        async_wait_and_barrier();

        // stage kk+64 into buf0 while computing buf1
        if (kk + 2 * BK < K_DIM) {
            stage_tile_async(xh,  rowBase0, kk + 2 * BK, As[0], tid);
            stage_tile_async(whT, colBase0, kk + 2 * BK, Bs[0], tid);
        }
        wmma_step(As[1], Bs[1], waveM, waveN, hi, l15, acc);
        async_wait_and_barrier();
    }

    // epilogue: out = scale*(acc + off*rowsum); C/D layout: VGPR r -> row r+8*hi, col l15
    #pragma unroll
    for (int nt = 0; nt < 2; ++nt) {
        const int gcol = colBase0 + waveN * 32 + nt * 16 + l15;
        const float sc = scale[gcol];
        const float of = offs[gcol];
        #pragma unroll
        for (int mt = 0; mt < 4; ++mt) {
            const int growBase = rowBase0 + waveM * 64 + mt * 16 + hi * 8;
            #pragma unroll
            for (int r = 0; r < 8; ++r) {
                const int grow = growBase + r;
                out[(size_t)grow * N_DIM + gcol] = sc * (acc[mt][nt][r] + of * rsum[grow]);
            }
        }
    }
}

// ---------------------------------------------------------------------------
// Fallback GEMM (small workspace): converts operands in-kernel (round-1 version)
// ---------------------------------------------------------------------------
__global__ __launch_bounds__(256) void dequant_gemm_fallback_kernel(
    const float* __restrict__ x, const int* __restrict__ w,
    const float* __restrict__ scale, const float* __restrict__ offs,
    const float* __restrict__ rsum, float* __restrict__ out) {

    __shared__ __align__(16) _Float16 Asf[BM * LDSH];
    __shared__ __align__(16) _Float16 Bsf[BN * LDSH];

    const int tid  = threadIdx.x;
    const int lane = tid & 31;
    const int wave = tid >> 5;
    const int hi   = lane >> 4;
    const int l15  = lane & 15;
    const int waveM = wave >> 2;
    const int waveN = wave & 3;
    const int rowBase0 = blockIdx.x * BM;
    const int colBase0 = blockIdx.y * BN;

    v8f acc[4][2];
    #pragma unroll
    for (int i = 0; i < 4; ++i)
        #pragma unroll
        for (int j = 0; j < 2; ++j)
            acc[i][j] = v8f{};

    for (int kk = 0; kk < K_DIM; kk += BK) {
        #pragma unroll
        for (int i = 0; i < 4; ++i) {
            const int idx = tid + i * 256;
            const int row = idx >> 3;
            const int c4  = idx & 7;
            const float4 xv = *(const float4*)(x + (size_t)(rowBase0 + row) * K_DIM + kk + c4 * 4);
            v4h h = { (_Float16)xv.x, (_Float16)xv.y, (_Float16)xv.z, (_Float16)xv.w };
            *(v4h*)(&Asf[row * LDSH + c4 * 4]) = h;
        }
        #pragma unroll
        for (int i = 0; i < 4; ++i) {
            const int idx = tid + i * 256;
            const int kr = idx >> 5;
            const int n4 = idx & 31;
            const int4 wv = *(const int4*)(w + (size_t)(kk + kr) * N_DIM + colBase0 + n4 * 4);
            Bsf[(n4 * 4 + 0) * LDSH + kr] = (_Float16)(float)wv.x;
            Bsf[(n4 * 4 + 1) * LDSH + kr] = (_Float16)(float)wv.y;
            Bsf[(n4 * 4 + 2) * LDSH + kr] = (_Float16)(float)wv.z;
            Bsf[(n4 * 4 + 3) * LDSH + kr] = (_Float16)(float)wv.w;
        }
        __syncthreads();
        wmma_step(Asf, Bsf, waveM, waveN, hi, l15, acc);
        __syncthreads();
    }

    #pragma unroll
    for (int nt = 0; nt < 2; ++nt) {
        const int gcol = colBase0 + waveN * 32 + nt * 16 + l15;
        const float sc = scale[gcol];
        const float of = offs[gcol];
        #pragma unroll
        for (int mt = 0; mt < 4; ++mt) {
            const int growBase = rowBase0 + waveM * 64 + mt * 16 + hi * 8;
            #pragma unroll
            for (int r = 0; r < 8; ++r) {
                const int grow = growBase + r;
                out[(size_t)grow * N_DIM + gcol] = sc * (acc[mt][nt][r] + of * rsum[grow]);
            }
        }
    }
}

// ---------------------------------------------------------------------------
extern "C" void kernel_launch(void* const* d_in, const int* in_sizes, int n_in,
                              void* d_out, int out_size, void* d_ws, size_t ws_size,
                              hipStream_t stream) {
    const float* x     = (const float*)d_in[0];
    const int*   w     = (const int*)d_in[1];
    const float* scale = (const float*)d_in[2];
    const float* offs  = (const float*)d_in[3];
    float* out = (float*)d_out;

    // workspace layout: [rsum: M floats][xh: M*K f16][whT: N*K f16]
    const size_t rsum_bytes = (size_t)M_DIM * sizeof(float);          // 16 KB
    const size_t xh_bytes   = (size_t)M_DIM * K_DIM * sizeof(_Float16);
    const size_t wh_bytes   = (size_t)N_DIM * K_DIM * sizeof(_Float16);
    float* rsum = (float*)d_ws;

    rowsum_kernel<<<M_DIM, 256, 0, stream>>>(x, rsum);

    dim3 grid(M_DIM / BM, N_DIM / BN);   // row-blocks fastest: weight slice reused from L2
    if (ws_size >= rsum_bytes + xh_bytes + wh_bytes) {
        _Float16* xh  = (_Float16*)((char*)d_ws + rsum_bytes);
        _Float16* whT = (_Float16*)((char*)d_ws + rsum_bytes + xh_bytes);
        convert_x_kernel<<<(M_DIM * (size_t)K_DIM) / 2048, 256, 0, stream>>>(x, xh);
        transpose_w_kernel<<<dim3(K_DIM / 64, N_DIM / 64), 256, 0, stream>>>(w, whT);
        dequant_gemm_f16_kernel<<<grid, 256, 0, stream>>>(xh, whT, scale, offs, rsum, out);
    } else {
        dequant_gemm_fallback_kernel<<<grid, 256, 0, stream>>>(x, w, scale, offs, rsum, out);
    }
}
